// GraphSageLayer_89326729822492
// MI455X (gfx1250) — compile-verified, matrix operand order checked by reading, not run on hardware
//
#include <hip/hip_runtime.h>
#include <cstdint>
#include <cstddef>

// Problem constants (from the reference)
#define BB   16
#define NN   1024
#define RR   128
#define KDIN 256
#define KATT 384
#define DOUTC 256

typedef __bf16 bf16;
typedef __attribute__((ext_vector_type(16))) __bf16 v16bf;
typedef __attribute__((ext_vector_type(8)))  __bf16 v8bf;
typedef __attribute__((ext_vector_type(8)))  float  v8f;

// ---------------------------------------------------------------------------
// WMMA helpers (CDNA5 wave32, V_WMMA_F32_16X16X32_BF16)
// ---------------------------------------------------------------------------
__device__ __forceinline__ v8f wmma_bf16(v16bf a, v16bf b, v8f c) {
  return __builtin_amdgcn_wmma_f32_16x16x32_bf16(false, a, false, b, (short)0, c,
                                                 false, false);
}

__device__ __forceinline__ v8f splat8(float x) {
  v8f z;
#pragma unroll
  for (int v = 0; v < 8; ++v) z[v] = x;
  return z;
}

// 16x32 bf16 A-fragment (also B as rows of B^T) from row-major, stride ld.
// Lane L (row=L&15, half=L>>4): elements 0..7 at K=half*8+{0..7},
// 8..15 at K=16+half*8+{0..7} -> two contiguous 16B chunks per lane.
__device__ __forceinline__ v16bf load_frag(const bf16* base, int ld) {
  const int lane = threadIdx.x & 31;
  const int row  = lane & 15;
  const int sub  = lane >> 4;
  const bf16* p = base + (size_t)row * ld + sub * 8;
  v8bf lo = *reinterpret_cast<const v8bf*>(p);
  v8bf hi = *reinterpret_cast<const v8bf*>(p + 16);
  return __builtin_shufflevector(lo, hi, 0, 1, 2, 3, 4, 5, 6, 7,
                                 8, 9, 10, 11, 12, 13, 14, 15);
}

__device__ __forceinline__ void store_tile_bf16(bf16* dst, int ld, v8f c, bool relu) {
  const int lane = threadIdx.x & 31;
  const int col = lane & 15;
  const int sub = lane >> 4;
#pragma unroll
  for (int v = 0; v < 8; ++v) {
    float x = c[v];
    if (relu) x = fmaxf(x, 0.0f);
    dst[(size_t)(v + sub * 8) * ld + col] = (bf16)x;
  }
}

// ---------------------------------------------------------------------------
// Utility kernels
// ---------------------------------------------------------------------------
__global__ void k_cvt(const float* __restrict__ s, bf16* __restrict__ d, size_t n) {
  size_t i = (size_t)blockIdx.x * blockDim.x + threadIdx.x;
  const size_t st = (size_t)gridDim.x * blockDim.x;
  for (; i < n; i += st) d[i] = (bf16)s[i];
}

// adj -> bf16 copy + bf16 transpose via 32x32 LDS tile (both sides coalesced)
__global__ __launch_bounds__(256) void k_adj(const float* __restrict__ a,
                                             bf16* __restrict__ ab,
                                             bf16* __restrict__ atb) {
  __shared__ bf16 tile[32][33];
  const int b = blockIdx.z;
  const int r0 = blockIdx.y * 32;
  const int c0 = blockIdx.x * 32;
  const int tc = threadIdx.x & 31;
  const int tr = threadIdx.x >> 5;  // 0..7
#pragma unroll
  for (int s = 0; s < 4; ++s) {
    const int r = tr + s * 8;
    const size_t idx = ((size_t)b << 20) | ((size_t)(r0 + r) << 10) | (size_t)(c0 + tc);
    const float v = a[idx];
    ab[idx] = (bf16)v;
    tile[r][tc] = (bf16)v;
  }
  __syncthreads();
#pragma unroll
  for (int s = 0; s < 4; ++s) {
    const int r = tr + s * 8;  // row of the transposed tile (= source col)
    atb[((size_t)b << 20) | ((size_t)(c0 + r) << 10) | (size_t)(r0 + tc)] = tile[tc][r];
  }
}

__global__ void k_zero(bf16* __restrict__ d, size_t n) {
  size_t i = (size_t)blockIdx.x * blockDim.x + threadIdx.x;
  const size_t st = (size_t)gridDim.x * blockDim.x;
  for (; i < n; i += st) d[i] = (bf16)0.0f;
}

// ---------------------------------------------------------------------------
// Projection: rep (stored transposed), node_addr (l2-normalized),
// node_id = relu(...) -> upd[:,128:256]. 3 waves, one weight matrix per wave.
// B-fragment loads are one-deep double-buffered so WMMA overlaps VMEM.
// ---------------------------------------------------------------------------
__global__ __launch_bounds__(96) void proj_kernel(
    const bf16* __restrict__ nodes_bf,
    const bf16* __restrict__ Wr, const bf16* __restrict__ Wa, const bf16* __restrict__ Wn,
    const float* __restrict__ br, const float* __restrict__ ba, const float* __restrict__ bn,
    bf16* __restrict__ rep_t, bf16* __restrict__ node_addr, bf16* __restrict__ upd) {
  const int wave = threadIdx.x >> 5;
  const int lane = threadIdx.x & 31;
  const int col = lane & 15;
  const int sub = lane >> 4;
  const int m0 = blockIdx.x * 16;

  v16bf af[8];
  const bf16* an = nodes_bf + (size_t)m0 * KDIN;
#pragma unroll
  for (int k = 0; k < 8; ++k) af[k] = load_frag(an + k * 32, KDIN);

  const bf16* W = (wave == 0) ? Wr : (wave == 1) ? Wa : Wn;
  const float* bias = (wave == 0) ? br : (wave == 1) ? ba : bn;

  v8f acc[8];
  v16bf wcur = load_frag(W, KDIN);
  for (int c = 0; c < 8; ++c) {
    acc[c] = splat8(bias[c * 16 + col]);
#pragma unroll
    for (int k = 0; k < 8; ++k) {
      const bf16* nb;
      if (k < 7)      nb = W + (size_t)(c * 16) * KDIN + (k + 1) * 32;
      else if (c < 7) nb = W + (size_t)((c + 1) * 16) * KDIN;
      else            nb = W;
      v16bf wnxt = load_frag(nb, KDIN);
      acc[c] = wmma_bf16(af[k], wcur, acc[c]);
      wcur = wnxt;
    }
  }

  const int b = m0 / NN;
  const int n0 = m0 % NN;
  if (wave == 0) {
#pragma unroll
    for (int c = 0; c < 8; ++c) {
#pragma unroll
      for (int v = 0; v < 8; ++v)
        rep_t[(size_t)b * RR * NN + (size_t)(c * 16 + col) * NN + n0 + v + sub * 8] =
            (bf16)acc[c][v];
    }
  } else if (wave == 1) {
    float rs[8];
#pragma unroll
    for (int v = 0; v < 8; ++v) {
      float s = 0.0f;
#pragma unroll
      for (int c = 0; c < 8; ++c) s += acc[c][v] * acc[c][v];
#pragma unroll
      for (int m = 1; m < 16; m <<= 1) s += __shfl_xor(s, m, 32);
      rs[v] = 1.0f / fmaxf(sqrtf(s), 1e-12f);
    }
#pragma unroll
    for (int c = 0; c < 8; ++c) {
#pragma unroll
      for (int v = 0; v < 8; ++v)
        node_addr[(size_t)(m0 + v + sub * 8) * RR + c * 16 + col] =
            (bf16)(acc[c][v] * rs[v]);
    }
  } else {
#pragma unroll
    for (int c = 0; c < 8; ++c) {
#pragma unroll
      for (int v = 0; v < 8; ++v)
        upd[(size_t)(m0 + v + sub * 8) * KATT + 128 + c * 16 + col] =
            (bf16)fmaxf(acc[c][v], 0.0f);
    }
  }
}

// ---------------------------------------------------------------------------
// Fused attention iteration (flash-style, see round-1 comments).
// New: mask tiles staged into LDS with global_load_async_to_lds_b128
// (ASYNCcnt path), all WMMA operand streams one-deep double-buffered.
// ---------------------------------------------------------------------------
__global__ __launch_bounds__(128) void attn_iter_kernel(
    const bf16* __restrict__ SA, const bf16* __restrict__ SB,
    const bf16* __restrict__ mask, const bf16* __restrict__ rep_t,
    const bf16* __restrict__ att_addr, const bf16* __restrict__ att_hid,
    const bf16* __restrict__ Watt, const float* __restrict__ batt,
    bf16* __restrict__ addr_out, bf16* __restrict__ hid_out,
    bf16* __restrict__ agg_out, int agg_off) {
  __shared__ __align__(16) bf16 stageP[4][16 * 32];   // P tile, C->A relayout
  __shared__ __align__(16) bf16 stageM[4][16 * 32];   // mask tile (async-to-LDS)
  __shared__ __align__(16) bf16 stageV[4][16 * 128];  // in_src, C->A relayout

  const int wave = threadIdx.x >> 5;
  const int lane = threadIdx.x & 31;
  const int col = lane & 15;
  const int sub = lane >> 4;
  const int b = blockIdx.y;
  const int i0 = (blockIdx.x * 4 + wave) * 16;
  const size_t rowbase = (size_t)b * NN + i0;

  v16bf qf[4];
  {
    const bf16* qb = SA + rowbase * RR;
#pragma unroll
    for (int k = 0; k < 4; ++k) qf[k] = load_frag(qb + k * 32, RR);
  }

  v8f acc[8];
#pragma unroll
  for (int c = 0; c < 8; ++c) acc[c] = splat8(0.0f);
  v8f asq = splat8(0.0f);

  const bf16* mrow = mask + (size_t)b * NN * NN + (size_t)i0 * NN;
  bf16* stp = stageP[wave];
  bf16* stm = stageM[wave];

  // Cooperative mask tile addressing: lane l covers row l>>1, cols (l&1)*16..+15
  const int mr = lane >> 1;
  const int mc = (lane & 1) * 16;
  const uint32_t mlds = (uint32_t)(uintptr_t)(stm + mr * 32 + mc);

  for (int jc = 0; jc < NN; jc += 32) {
    // --- async mask tile: global -> LDS, 2 x b128 per lane, no VGPR round-trip
    {
      const bf16* msrc = mrow + (size_t)mr * NN + jc + mc;
      asm volatile("global_load_async_to_lds_b128 %0, %1, off"
                   :: "v"(mlds), "v"(msrc) : "memory");
      asm volatile("global_load_async_to_lds_b128 %0, %1, off offset:16"
                   :: "v"(mlds), "v"(msrc) : "memory");
    }

    // --- S = Q @ K^T, two 16-col tiles, B loads pipelined one deep ---
    const bf16* kb = SB + ((size_t)b * NN + jc) * RR;
    v8f s0 = splat8(0.0f), s1 = splat8(0.0f);
    v16bf bcur = load_frag(kb, RR);
#pragma unroll
    for (int k = 0; k < 4; ++k) {
      v16bf bnxt = load_frag(kb + (k < 3 ? (k + 1) * 32 : 16 * RR), RR);
      s0 = wmma_bf16(qf[k], bcur, s0);
      bcur = bnxt;
    }
#pragma unroll
    for (int k = 0; k < 4; ++k) {
      v16bf bnxt = (k < 3) ? load_frag(kb + 16 * RR + (k + 1) * 32, RR) : bcur;
      s1 = wmma_bf16(qf[k], bcur, s1);
      bcur = bnxt;
    }

    // --- wait for the mask tile, then P = exp(S)*mask, stage P, sum P^2 ---
    asm volatile("s_wait_asynccnt 0x0" ::: "memory");
#pragma unroll
    for (int v = 0; v < 8; ++v) {
      const int il = v + sub * 8;
      const float m0v = (float)stm[il * 32 + col];
      const float m1v = (float)stm[il * 32 + 16 + col];
      const float p0 = __expf(s0[v]) * m0v;
      const float p1 = __expf(s1[v]) * m1v;
      asq[v] += p0 * p0 + p1 * p1;
      stp[il * 32 + col] = (bf16)p0;
      stp[il * 32 + 16 + col] = (bf16)p1;
    }
    asm volatile("s_wait_dscnt 0" ::: "memory");
    v16bf pf = load_frag(stp, 32);

    // --- acc += P(16x32) @ rep chunk, B loads (rep_t) pipelined one deep ---
    const bf16* rbase = rep_t + (size_t)b * RR * NN + jc;
    v16bf rcur = load_frag(rbase, NN);
#pragma unroll
    for (int c = 0; c < 8; ++c) {
      v16bf rnxt = (c < 7) ? load_frag(rbase + (size_t)((c + 1) * 16) * NN, NN) : rcur;
      acc[c] = wmma_bf16(pf, rcur, acc[c]);
      rcur = rnxt;
    }
    // All DS reads (mask + P frag) must retire before next chunk's async/DS writes
    asm volatile("s_wait_dscnt 0" ::: "memory");
  }

  // --- l2 row normalization (reduce across the 16 lanes of each half) ---
  float rs[8];
#pragma unroll
  for (int v = 0; v < 8; ++v) {
    float s = asq[v];
#pragma unroll
    for (int m = 1; m < 16; m <<= 1) s += __shfl_xor(s, m, 32);
    rs[v] = 1.0f / fmaxf(sqrtf(s), 1e-12f);
  }

  // --- stage in_src; build concat(in_src, addr, hidden) A-fragments ---
  bf16* stv = stageV[wave];
#pragma unroll
  for (int c = 0; c < 8; ++c) {
#pragma unroll
    for (int v = 0; v < 8; ++v)
      stv[(v + sub * 8) * 128 + c * 16 + col] = (bf16)(acc[c][v] * rs[v]);
  }
  asm volatile("s_wait_dscnt 0" ::: "memory");

  v16bf af[12];
#pragma unroll
  for (int k = 0; k < 4; ++k) af[k] = load_frag(stv + k * 32, 128);
  {
    const bf16* ab = att_addr + rowbase * RR;
    const bf16* hb = att_hid + rowbase * RR;
#pragma unroll
    for (int k = 0; k < 4; ++k) af[4 + k] = load_frag(ab + k * 32, RR);
#pragma unroll
    for (int k = 0; k < 4; ++k) af[8 + k] = load_frag(hb + k * 32, RR);
  }

  // --- att(): 24 tiles x K=384, W fragments pipelined across tile edges ---
  v16bf wcur = load_frag(Watt, KATT);
  for (int o = 0; o < 24; ++o) {
    v8f c = splat8(batt[o * 16 + col]);
    const bf16* wb = Watt + (size_t)(o * 16) * KATT;
#pragma unroll
    for (int k = 0; k < 12; ++k) {
      const bf16* nb;
      if (k < 11)      nb = wb + (k + 1) * 32;
      else if (o < 23) nb = Watt + (size_t)((o + 1) * 16) * KATT;
      else             nb = wb;
      v16bf wnxt = load_frag(nb, KATT);
      c = wmma_bf16(af[k], wcur, c);
      wcur = wnxt;
    }
    if (o < 8) {
      store_tile_bf16(agg_out + rowbase * KATT + agg_off + o * 16, KATT, c, true);
    } else if (o < 16) {
      store_tile_bf16(addr_out + rowbase * RR + (o - 8) * 16, RR, c, false);
    } else {
      store_tile_bf16(hid_out + rowbase * RR + (o - 16) * 16, RR, c, true);
    }
  }
}

// ---------------------------------------------------------------------------
// Final layer: out = relu(upd @ W_upd^T + b_upd), K=384, f32 output
// ---------------------------------------------------------------------------
__global__ __launch_bounds__(128) void final_kernel(
    const bf16* __restrict__ upd, const bf16* __restrict__ Wu,
    const float* __restrict__ bu, float* __restrict__ out) {
  const int wave = threadIdx.x >> 5;
  const int lane = threadIdx.x & 31;
  const int col = lane & 15;
  const int sub = lane >> 4;
  const int m0 = blockIdx.x * 16;

  v16bf af[12];
  const bf16* ub = upd + (size_t)m0 * KATT;
#pragma unroll
  for (int k = 0; k < 12; ++k) af[k] = load_frag(ub + k * 32, KATT);

  v16bf wcur = load_frag(Wu + (size_t)(wave * 4 * 16) * KATT, KATT);
#pragma unroll
  for (int t = 0; t < 4; ++t) {
    const int o = wave * 4 + t;
    v8f c = splat8(bu[o * 16 + col]);
    const bf16* wb = Wu + (size_t)(o * 16) * KATT;
#pragma unroll
    for (int k = 0; k < 12; ++k) {
      const bf16* nb;
      if (k < 11)     nb = wb + (k + 1) * 32;
      else if (t < 3) nb = Wu + (size_t)((o + 1) * 16) * KATT;
      else            nb = wb;
      v16bf wnxt = load_frag(nb, KATT);
      c = wmma_bf16(af[k], wcur, c);
      wcur = wnxt;
    }
#pragma unroll
    for (int v = 0; v < 8; ++v)
      out[(size_t)(m0 + v + sub * 8) * DOUTC + o * 16 + col] = fmaxf(c[v], 0.0f);
  }
}

// ---------------------------------------------------------------------------
// Host-side orchestration (graph-capture safe: kernel launches only).
// Workspace ~118 MB: bf16 nodes/weights/adj/adj^T + activation ping-pong.
// ---------------------------------------------------------------------------
extern "C" void kernel_launch(void* const* d_in, const int* in_sizes, int n_in,
                              void* d_out, int out_size, void* d_ws, size_t ws_size,
                              hipStream_t stream) {
  (void)in_sizes; (void)n_in; (void)out_size; (void)ws_size;
  const float* nodes  = (const float*)d_in[0];
  const float* adj    = (const float*)d_in[1];
  const float* W_rep  = (const float*)d_in[2];
  const float* b_rep  = (const float*)d_in[3];
  const float* W_addr = (const float*)d_in[4];
  const float* b_addr = (const float*)d_in[5];
  const float* W_nrep = (const float*)d_in[6];
  const float* b_nrep = (const float*)d_in[7];
  const float* W_att  = (const float*)d_in[8];
  const float* b_att  = (const float*)d_in[9];
  const float* W_upd  = (const float*)d_in[10];
  const float* b_upd  = (const float*)d_in[11];
  float* out = (float*)d_out;

  char* cur = (char*)d_ws;
  auto take = [&](size_t bytes) -> char* {
    char* p = cur;
    cur += (bytes + 255) & ~(size_t)255;
    return p;
  };

  bf16* nodes_bf = (bf16*)take((size_t)BB * NN * KDIN * 2);
  bf16* Wr_bf    = (bf16*)take((size_t)RR * KDIN * 2);
  bf16* Wa_bf    = (bf16*)take((size_t)RR * KDIN * 2);
  bf16* Wn_bf    = (bf16*)take((size_t)RR * KDIN * 2);
  bf16* Watt_bf  = (bf16*)take((size_t)KATT * KATT * 2);
  bf16* Wupd_bf  = (bf16*)take((size_t)DOUTC * KATT * 2);
  bf16* naddr    = (bf16*)take((size_t)BB * NN * RR * 2);
  bf16* rep_t    = (bf16*)take((size_t)BB * RR * NN * 2);
  bf16* zbuf     = (bf16*)take((size_t)BB * NN * RR * 2);   // stays zero
  bf16* b1a      = (bf16*)take((size_t)BB * NN * RR * 2);
  bf16* b1h      = (bf16*)take((size_t)BB * NN * RR * 2);
  bf16* b2a      = (bf16*)take((size_t)BB * NN * RR * 2);
  bf16* b2h      = (bf16*)take((size_t)BB * NN * RR * 2);
  bf16* updb     = (bf16*)take((size_t)BB * NN * KATT * 2);
  bf16* adj_bf   = (bf16*)take((size_t)BB * NN * NN * 2);
  bf16* adjT_bf  = (bf16*)take((size_t)BB * NN * NN * 2);

  const dim3 cb(256);
  k_cvt<<<2048, cb, 0, stream>>>(nodes, nodes_bf, (size_t)BB * NN * KDIN);
  k_cvt<<<64, cb, 0, stream>>>(W_rep, Wr_bf, (size_t)RR * KDIN);
  k_cvt<<<64, cb, 0, stream>>>(W_addr, Wa_bf, (size_t)RR * KDIN);
  k_cvt<<<64, cb, 0, stream>>>(W_nrep, Wn_bf, (size_t)RR * KDIN);
  k_cvt<<<256, cb, 0, stream>>>(W_att, Watt_bf, (size_t)KATT * KATT);
  k_cvt<<<192, cb, 0, stream>>>(W_upd, Wupd_bf, (size_t)DOUTC * KATT);
  k_adj<<<dim3(NN / 32, NN / 32, BB), cb, 0, stream>>>(adj, adj_bf, adjT_bf);
  k_zero<<<512, cb, 0, stream>>>(zbuf, (size_t)BB * NN * RR);

  proj_kernel<<<dim3(BB * NN / 16), dim3(96), 0, stream>>>(
      nodes_bf, Wr_bf, Wa_bf, Wn_bf, b_rep, b_addr, b_nrep, rep_t, naddr, updb);

  const dim3 ag(NN / 64, BB), abk(128);
  // in-direction: S rows = addr (iter), keys = node_addr (fixed), mask = adj
  attn_iter_kernel<<<ag, abk, 0, stream>>>(zbuf, naddr, adj_bf, rep_t, zbuf, zbuf,
                                           Watt_bf, b_att, b1a, b1h, updb, 0);
  attn_iter_kernel<<<ag, abk, 0, stream>>>(b1a, naddr, adj_bf, rep_t, b1a, b1h,
                                           Watt_bf, b_att, b2a, b2h, updb, 0);
  attn_iter_kernel<<<ag, abk, 0, stream>>>(b2a, naddr, adj_bf, rep_t, b2a, b2h,
                                           Watt_bf, b_att, b1a, b1h, updb, 0);
  // out-direction: rows = j with Q = node_addr (fixed), keys = addr, mask = adj^T
  attn_iter_kernel<<<ag, abk, 0, stream>>>(naddr, zbuf, adjT_bf, rep_t, zbuf, zbuf,
                                           Watt_bf, b_att, b1a, b1h, updb, 256);
  attn_iter_kernel<<<ag, abk, 0, stream>>>(naddr, b1a, adjT_bf, rep_t, b1a, b1h,
                                           Watt_bf, b_att, b2a, b2h, updb, 256);
  attn_iter_kernel<<<ag, abk, 0, stream>>>(naddr, b2a, adjT_bf, rep_t, b2a, b2h,
                                           Watt_bf, b_att, b1a, b1h, updb, 256);

  final_kernel<<<dim3(BB * NN / 16), dim3(128), 0, stream>>>(updb, Wupd_bf, b_upd, out);
}
